// EnvEmbedding_51539607552205
// MI455X (gfx1250) — matrix operand order repaired and checked by reading, not compiled, not run on hardware
//
#include <hip/hip_runtime.h>
#include <stdint.h>

typedef __attribute__((ext_vector_type(16))) __bf16 v16bf;
typedef __attribute__((ext_vector_type(8)))  float  v8f;

#define D        512
#define KCB      8192
#define MBLK     128      // rows per block
#define NCHUNK   64       // codes per chunk
#define XSTRIDE  520      // 512 + 8 pad (elems) -> conflict-free ds_load_b128
#define NTHREADS 256

union Frag { int4 q[2]; v16bf v; };

static __device__ __forceinline__ uint32_t bf16rne(float f) {
    uint32_t u = __float_as_uint(f);
    return (u + 0x7FFFu + ((u >> 16) & 1u)) >> 16;   // round-to-nearest-even
}

// ---------------------------------------------------------------------------
// Kernel 1: codebook fp32 -> bf16 hi/lo planes + per-code squared norms.
// One wave per code (8 waves/block). Coalesced loads, wave-shuffle reduction.
// ---------------------------------------------------------------------------
__global__ __launch_bounds__(NTHREADS)
void vq_prep_codebook(const float* __restrict__ cb,
                      uint16_t* __restrict__ cb_hi,
                      uint16_t* __restrict__ cb_lo,
                      float* __restrict__ cbnorm) {
    const int w    = threadIdx.x >> 5;
    const int lane = threadIdx.x & 31;
    const int code = blockIdx.x * 8 + w;
    const float* row = cb + (size_t)code * D;
    uint16_t* hrow = cb_hi + (size_t)code * D;
    uint16_t* lrow = cb_lo + (size_t)code * D;
    float acc = 0.0f;
#pragma unroll
    for (int i = 0; i < D / 32; ++i) {
        int d = lane + i * 32;
        float f = row[d];
        uint32_t h = bf16rne(f);
        float hf = __uint_as_float(h << 16);
        uint32_t l = bf16rne(f - hf);
        hrow[d] = (uint16_t)h;
        lrow[d] = (uint16_t)l;
        acc += f * f;
    }
#pragma unroll
    for (int off = 16; off; off >>= 1) acc += __shfl_xor(acc, off, 32);
    if (lane == 0) cbnorm[code] = acc;
}

// ---------------------------------------------------------------------------
// Kernel 2: fused GEMM (bf16 hi/lo split, f32 accum) + running argmin.
// Block = 128 rows x all 8192 codes. 8 waves: 4 along M x 2 along N.
// Wave computes a 32x32 tile (2x2 WMMA tiles) per 64-code chunk.
// ---------------------------------------------------------------------------
__global__ __launch_bounds__(NTHREADS)
void vq_argmin(const float* __restrict__ x,
               const uint16_t* __restrict__ cb_hi,
               const uint16_t* __restrict__ cb_lo,
               const float* __restrict__ cbnorm,
               int* __restrict__ out) {
    __shared__ __align__(16) char smem[2 * MBLK * XSTRIDE * 2];   // 266,240 B
    uint16_t* xhi = (uint16_t*)smem;
    uint16_t* xlo = xhi + MBLK * XSTRIDE;

    const int tid  = threadIdx.x;
    const int lane = tid & 31;
    const int w    = tid >> 5;
    const int mw   = w >> 1;        // 0..3  (M wave)
    const int nw   = w & 1;         // 0..1  (N wave)
    const int lh   = lane >> 4;     // lane half
    const int ln   = lane & 15;

    // ---- Phase 0: load x tile (128 x 512 f32), split to bf16 hi/lo in LDS ----
    {
        const float4* x4 = (const float4*)(x + (size_t)blockIdx.x * MBLK * D);
#pragma unroll 4
        for (int i = 0; i < (MBLK * D / 4) / NTHREADS; ++i) {
            int f4  = i * NTHREADS + tid;
            int row = f4 >> 7;              // 128 float4 per row
            int d   = (f4 & 127) << 2;
            float4 t = x4[f4];
            uint32_t h0 = bf16rne(t.x), h1 = bf16rne(t.y),
                     h2 = bf16rne(t.z), h3 = bf16rne(t.w);
            uint32_t l0 = bf16rne(t.x - __uint_as_float(h0 << 16));
            uint32_t l1 = bf16rne(t.y - __uint_as_float(h1 << 16));
            uint32_t l2 = bf16rne(t.z - __uint_as_float(h2 << 16));
            uint32_t l3 = bf16rne(t.w - __uint_as_float(h3 << 16));
            int o = row * XSTRIDE + d;
            *(uint2*)(xhi + o) = make_uint2(h0 | (h1 << 16), h2 | (h3 << 16));
            *(uint2*)(xlo + o) = make_uint2(l0 | (l1 << 16), l2 | (l3 << 16));
        }
    }
    __syncthreads();

    // ---- running per-lane argmin state: 16 (row-slot) entries ----
    float best[16];
    int   bidx[16];
#pragma unroll
    for (int s = 0; s < 16; ++s) { best[s] = 3.4e38f; bidx[s] = 0; }

    // precomputed fragment base offsets
    const int arow0 = mw * 32 + ln;                 // mt adds +16
    const int abase = arow0 * XSTRIDE + lh * 8;     // + d0 ; +16 for elems 8..15
    const int bcode0 = nw * 32 + ln;                // mt: +nt*16 ; chunk adds cbase

#pragma unroll 1
    for (int c = 0; c < KCB / NCHUNK; ++c) {
        const int cbase = c * NCHUNK;
        v8f acc[2][2];
        const v8f zf = {0.f, 0.f, 0.f, 0.f, 0.f, 0.f, 0.f, 0.f};
#pragma unroll
        for (int mt = 0; mt < 2; ++mt)
#pragma unroll
            for (int nt = 0; nt < 2; ++nt) acc[mt][nt] = zf;

#pragma unroll 2
        for (int d0 = 0; d0 < D; d0 += 32) {
            Frag ahi[2], alo[2], bhi[2], blo[2];
#pragma unroll
            for (int mt = 0; mt < 2; ++mt) {
                int o = abase + mt * 16 * XSTRIDE + d0;
                ahi[mt].q[0] = *(const int4*)(xhi + o);
                ahi[mt].q[1] = *(const int4*)(xhi + o + 16);
                alo[mt].q[0] = *(const int4*)(xlo + o);
                alo[mt].q[1] = *(const int4*)(xlo + o + 16);
            }
#pragma unroll
            for (int nt = 0; nt < 2; ++nt) {
                int code = cbase + bcode0 + nt * 16;
                int g = code * D + d0 + lh * 8;     // k = 8h+s (s<8), 16+8h+(s-8) (s>=8)
                bhi[nt].q[0] = *(const int4*)(cb_hi + g);
                bhi[nt].q[1] = *(const int4*)(cb_hi + g + 16);
                blo[nt].q[0] = *(const int4*)(cb_lo + g);
                blo[nt].q[1] = *(const int4*)(cb_lo + g + 16);
            }
#pragma unroll
            for (int mt = 0; mt < 2; ++mt)
#pragma unroll
                for (int nt = 0; nt < 2; ++nt) {
                    acc[mt][nt] = __builtin_amdgcn_wmma_f32_16x16x32_bf16(
                        false, ahi[mt].v, false, bhi[nt].v, (short)0, acc[mt][nt], false, false);
                    acc[mt][nt] = __builtin_amdgcn_wmma_f32_16x16x32_bf16(
                        false, ahi[mt].v, false, blo[nt].v, (short)0, acc[mt][nt], false, false);
                    acc[mt][nt] = __builtin_amdgcn_wmma_f32_16x16x32_bf16(
                        false, alo[mt].v, false, bhi[nt].v, (short)0, acc[mt][nt], false, false);
                }
        }

        // dist = ||c||^2 - 2 * dot ; strict < keeps first (smallest) code on ties
#pragma unroll
        for (int nt = 0; nt < 2; ++nt) {
            int code = cbase + bcode0 + nt * 16;
            float cn = cbnorm[code];
#pragma unroll
            for (int mt = 0; mt < 2; ++mt)
#pragma unroll
                for (int v = 0; v < 8; ++v) {
                    float dv = cn - 2.0f * acc[mt][nt][v];
                    int s = mt * 8 + v;
                    if (dv < best[s]) { best[s] = dv; bidx[s] = code; }
                }
        }
    }

    // ---- final cross-lane / cross-wave argmin via LDS (aliases x tile) ----
    __syncthreads();
    float* rv = (float*)smem;                   // [128][32]
    int*   ri = (int*)(smem + MBLK * 32 * 4);   // [128][32]
    const int col = nw * 16 + ln;
#pragma unroll
    for (int mt = 0; mt < 2; ++mt)
#pragma unroll
        for (int v = 0; v < 8; ++v) {
            int row = mw * 32 + mt * 16 + lh * 8 + v;
            int s = mt * 8 + v;
            rv[row * 32 + col] = best[s];
            ri[row * 32 + col] = bidx[s];
        }
    __syncthreads();

    if (tid < MBLK) {
        int row = tid;
        float bv = rv[row * 32];
        int   bi = ri[row * 32];
#pragma unroll 4
        for (int j = 1; j < 32; ++j) {
            float v = rv[row * 32 + j];
            int   i = ri[row * 32 + j];
            if (v < bv || (v == bv && i < bi)) { bv = v; bi = i; }
        }
        out[blockIdx.x * MBLK + row] = bi;
    }
}

// ---------------------------------------------------------------------------
extern "C" void kernel_launch(void* const* d_in, const int* in_sizes, int n_in,
                              void* d_out, int out_size, void* d_ws, size_t ws_size,
                              hipStream_t stream) {
    const float* x  = (const float*)d_in[0];        // 64*512*512 f32
    const float* cb = (const float*)d_in[1];        // 8192*512 f32

    // workspace: cb_hi (8 MB) | cb_lo (8 MB) | cbnorm (32 KB)
    uint16_t* cb_hi  = (uint16_t*)d_ws;
    uint16_t* cb_lo  = cb_hi + (size_t)KCB * D;
    float*    cbnorm = (float*)(cb_lo + (size_t)KCB * D);

    vq_prep_codebook<<<KCB / 8, NTHREADS, 0, stream>>>(cb, cb_hi, cb_lo, cbnorm);

    const int nrows = 64 * 512;                     // 32768 rows of D=512
    vq_argmin<<<nrows / MBLK, NTHREADS, 0, stream>>>(x, cb_hi, cb_lo, cbnorm,
                                                     (int*)d_out);
}
